// StatePotentialModel_47845935677913
// MI455X (gfx1250) — matrix operand order repaired and checked by reading, not compiled, not run on hardware
//
#include <hip/hip_runtime.h>
#include <hip/hip_bf16.h>
#include <math.h>

// Problem constants (from reference): B,N,F,D,E,M,L
constexpr int cB = 4, cN = 4096, cF = 32, cD = 128, cE = 65536, cM = 1024, cL = 6;
constexpr int cROWS = cB * cN;      // 16384 node rows
constexpr int cCHUNK = 128 * 128;   // one swizzled weight chunk: 16384 f16 = 32 KB

typedef __attribute__((ext_vector_type(16))) _Float16 v16h;
typedef __attribute__((ext_vector_type(8)))  float    v8f;
typedef unsigned int  u32x4 __attribute__((ext_vector_type(4)));
typedef int           i32x4 __attribute__((ext_vector_type(4)));
typedef int           i32x8 __attribute__((ext_vector_type(8)));

// fast silu: x * rcp(1+exp(-x)) -> v_exp_f32 + v_add + v_rcp_f32 + v_mul
__device__ __forceinline__ float silu_f(float x) {
  return x * __builtin_amdgcn_rcpf(1.0f + __expf(-x));
}

// ---------------- WMMA fragment helpers ----------------
// A-fragment (16x32 f16, row-major in LDS with given stride).
// Layout: lanes 0-15 hold row M=lane, K {kb..kb+15} with kb=0;
// lanes 16-31 same rows with kb=16 (matches bswz B order).  One contiguous 32B group.
__device__ __forceinline__ v16h frag_a(const _Float16* a, int stride, int k0, int lane) {
  int row = lane & 15;
  int kb  = k0 + ((lane & 16) ? 16 : 0);
  const _Float16* p = a + row * stride + kb;
  v16h r;
#pragma unroll
  for (int i = 0; i < 8; ++i) { r[i] = p[i]; r[8 + i] = p[8 + i]; }
  return r;
}

// B tiles live in LDS pre-swizzled to fragment-major order so each lane reads
// 16 contiguous (32B-aligned) halfwords -> vector DS load.
// Logical B element (k, n) of a K x 128 tile lives at:
//   kt = k>>5; lane = (n&15) | (k&16); i = k&15; wcol = n>>4;
//   idx = (((kt*8)+wcol)*32 + lane)*16 + i
__device__ __forceinline__ int bswz(int k, int n) {
  int kt = k >> 5;
  int lane = (n & 15) | (k & 16);
  int i = k & 15;
  int wcol = n >> 4;
  return (((kt * 8) + wcol) * 32 + lane) * 16 + i;
}
__device__ __forceinline__ v16h frag_bs(const _Float16* bS, int kt, int w, int lane) {
  return *(const v16h*)(bS + ((((kt * 8) + w) * 32) + lane) * 16);
}
// stage a K x 128 f32 weight tile into swizzled f16 LDS (all 256 threads)
__device__ __forceinline__ void stage_b(_Float16* bS, const float* W, int K, int tid) {
  for (int i = tid; i < K * cD; i += 256) {
    int k = i >> 7, n = i & 127;
    bS[bswz(k, n)] = (_Float16)W[i];
  }
}

#define WMMA_F16(c, a, b) \
  (c) = __builtin_amdgcn_wmma_f32_16x16x32_f16(false, (a), false, (b), (short)0, (c), false, false)

// ---------------- Tensor Data Mover: 32KB contiguous global->LDS ----------------
// 1-D descriptor per CDNA5 ISA:
//   group0: count=1 | lds_addr | global_addr[56:0] | type=2
//   group1: data_size=1(2B), tensor_dim0=tile_dim0=16384, tensor_dim1=tile_dim1=1,
//           tensor_dim0_stride=16384
// Issued by one wave; completion via s_wait_tensorcnt 0, then workgroup barrier.
__device__ __forceinline__ void tdm_load_chunk(_Float16* ldsDst, const _Float16* gsrc) {
  unsigned int lds = (unsigned int)(unsigned long long)(uintptr_t)ldsDst; // flat LDS aperture: addr[31:0] = LDS offset
  unsigned long long ga = (unsigned long long)(uintptr_t)gsrc;
  u32x4 g0;
  g0[0] = 1u;                                        // count=1, user descriptor
  g0[1] = lds;                                       // lds_addr (bytes)
  g0[2] = (unsigned int)(ga & 0xffffffffull);        // global_addr[31:0]
  g0[3] = (unsigned int)((ga >> 32) & 0x01ffffffull) // global_addr[56:32]
          | (2u << 30);                              // type = 2 ("image")
  i32x8 g1;
  g1[0] = 1 << 16;                      // workgroup_mask=0, data_size=1 (2 bytes)
  g1[1] = (int)(((unsigned)cCHUNK & 0xffffu) << 16);        // tensor_dim0[15:0] -> bits 63:48
  g1[2] = (int)((((unsigned)cCHUNK >> 16) & 0xffffu)        // tensor_dim0[31:16]
          | (1u << 16));                                    // tensor_dim1 = 1
  g1[3] = (int)(((unsigned)cCHUNK & 0xffffu) << 16);        // tile_dim0 = 16384 -> bits 127:112
  g1[4] = 1;                            // tile_dim1 = 1, tile_dim2 = 0
  g1[5] = cCHUNK;                       // tensor_dim0_stride[31:0]
  g1[6] = 0;
  g1[7] = 0;
  i32x4 z4 = {0, 0, 0, 0};
#if __clang_major__ >= 23
  i32x8 z8 = {0, 0, 0, 0, 0, 0, 0, 0};
  __builtin_amdgcn_tensor_load_to_lds(g0, g1, z4, z4, z8, 0);
#else
  __builtin_amdgcn_tensor_load_to_lds(g0, g1, z4, z4, 0);
#endif
}

// -------- one-shot weight pack: Wg (12 chunks) + Wd1 (4) + Wd2 (1) -> swizzled f16 --------
__global__ void k_prep(const float* __restrict__ Wg, const float* __restrict__ Wd1,
                       const float* __restrict__ Wd2, _Float16* __restrict__ wpack) {
  int gid = blockIdx.x * blockDim.x + threadIdx.x;
  if (gid >= 17 * cCHUNK) return;
  int c = gid >> 14, iw = gid & (cCHUNK - 1);
  int k = iw >> 7, n = iw & 127;
  const float* src = (c < 12) ? (Wg + (size_t)c * cCHUNK)
                   : (c < 16) ? (Wd1 + (size_t)(c - 12) * cCHUNK)
                              : Wd2;
  wpack[(size_t)c * cCHUNK + bswz(k, n)] = (_Float16)src[iw];
}

// -------- time embedding: t (B,) -> t_emb (B,D) --------
__global__ void k_temb(const float* __restrict__ t, const float* __restrict__ Wt1,
                       const float* __restrict__ bt1, const float* __restrict__ Wt2,
                       const float* __restrict__ bt2, float* __restrict__ temb) {
  __shared__ float te[cB * 16];
  __shared__ float h1[cB * cD];
  int tid = threadIdx.x;
  if (tid < cB * 16) {
    int b = tid >> 4, j = tid & 15, jj = j & 7;
    float freq = __expf((float)jj * (-logf(10000.0f) / 8.0f));
    float arg  = t[b] * freq;
    te[tid] = (j < 8) ? __sinf(arg) : __cosf(arg);
  }
  __syncthreads();
  if (tid < cB * cD) {
    int b = tid / cD, d = tid % cD;
    float acc = bt1[d];
    for (int j = 0; j < 16; ++j) acc += te[b * 16 + j] * Wt1[j * cD + d];
    h1[tid] = silu_f(acc);
  }
  __syncthreads();
  if (tid < cB * cD) {
    int b = tid / cD, d = tid % cD;
    float acc = bt2[d];
    for (int k = 0; k < cD; ++k) acc += h1[b * cD + k] * Wt2[k * cD + d];
    temb[tid] = acc;
  }
}

// -------- encoder: h = silu(x@We1+be1)@We2 + be2 + temb[b] --------
__global__ void k_encoder(const float* __restrict__ x, const float* __restrict__ We1,
                          const float* __restrict__ be1, const float* __restrict__ We2,
                          const float* __restrict__ be2, const float* __restrict__ temb,
                          float* __restrict__ h) {
  __shared__ _Float16 aX[16 * cF];   // 1 KB
  __shared__ _Float16 w1[cF * cD];   // 8 KB  (swizzled)
  __shared__ _Float16 w2[cD * cD];   // 32 KB (swizzled)
  __shared__ _Float16 tA[16 * cD];   // 4 KB
  int tid = threadIdx.x, lane = tid & 31, w = tid >> 5;
  int row0 = blockIdx.x * 16;

  for (int i = tid; i < 16 * cF; i += 256) {
    int m = i / cF, k = i % cF;
    aX[i] = (_Float16)x[(size_t)(row0 + m) * cF + k];
  }
  stage_b(w1, We1, cF, tid);
  stage_b(w2, We2, cD, tid);
  __syncthreads();

  v8f c1 = {};
  { v16h a = frag_a(aX, cF, 0, lane); v16h b = frag_bs(w1, 0, w, lane); WMMA_F16(c1, a, b); }
  {
    int n = w * 16 + (lane & 15), mb = (lane & 16) ? 8 : 0;
#pragma unroll
    for (int r = 0; r < 8; ++r)
      tA[(mb + r) * cD + n] = (_Float16)silu_f(c1[r] + be1[n]);
  }
  __syncthreads();

  v8f c2 = {};
#pragma unroll
  for (int kt = 0; kt < 4; ++kt) {
    v16h a = frag_a(tA, cD, kt * 32, lane);
    v16h b = frag_bs(w2, kt, w, lane);
    WMMA_F16(c2, a, b);
  }
  {
    int n = w * 16 + (lane & 15), mb = (lane & 16) ? 8 : 0;
#pragma unroll
    for (int r = 0; r < 8; ++r) {
      int gr = row0 + mb + r;
      int bb = gr >> 12;  // / cN
      h[(size_t)gr * cD + n] = c2[r] + be2[n] + temb[bb * cD + n];
    }
  }
}

// -------- degree histogram --------
__global__ void k_deg(const int* __restrict__ ei, float* __restrict__ deg) {
  int e = blockIdx.x * blockDim.x + threadIdx.x;
  if (e < cE) atomicAdd(&deg[ei[cE + e]], 1.0f);
}

// -------- edge scatter: agg[b][dst] += h[b][src] --------
__global__ void k_scatter(const int* __restrict__ ei, const float* __restrict__ h,
                          float* __restrict__ agg) {
  int gid = blockIdx.x * blockDim.x + threadIdx.x;  // cE * 32 threads
  int e = gid >> 5;
  if (e >= cE) return;
  int d4 = (gid & 31) * 4;
  int s = ei[e], d = ei[cE + e];
#pragma unroll
  for (int b = 0; b < cB; ++b) {
    const float* hp = &h[((size_t)b * cN + s) * cD + d4];
    if (b + 1 < cB) __builtin_prefetch(&h[((size_t)(b + 1) * cN + s) * cD + d4], 0, 0);
    const float4 v = *(const float4*)hp;
    float* ap = &agg[((size_t)b * cN + d) * cD + d4];
    atomicAdd(ap + 0, v.x); atomicAdd(ap + 1, v.y);
    atomicAdd(ap + 2, v.z); atomicAdd(ap + 3, v.w);
  }
}

// -------- GNN layer: h = LN(h + silu([h|agg/deg]@Wg + bg)) --------
// Weight chunks arrive via Tensor Data Mover (pre-swizzled f16 in global).
__global__ void k_layer(const float* __restrict__ h, const float* __restrict__ agg,
                        const float* __restrict__ deg, const _Float16* __restrict__ wgp,
                        const float* __restrict__ bg, const float* __restrict__ lg,
                        const float* __restrict__ lb, float* __restrict__ hout) {
  __shared__ _Float16 aT[16 * 256];  // 8 KB  (A = [h | agg/deg])
  __shared__ _Float16 bT[cCHUNK];    // 32 KB (half of Wg at a time, swizzled, TDM-filled)
  __shared__ float    yT[16 * cD];   // 8 KB
  __shared__ float    mu[16], rs[16];
  int tid = threadIdx.x, lane = tid & 31, w = tid >> 5;
  int row0 = blockIdx.x * 16;

  for (int i = tid; i < 16 * 256; i += 256) {
    int m = i >> 8, k = i & 255;
    int gr = row0 + m;
    float v;
    if (k < cD) v = h[(size_t)gr * cD + k];
    else {
      int node = gr & (cN - 1);
      v = agg[(size_t)gr * cD + (k - cD)] * __builtin_amdgcn_rcpf(fmaxf(deg[node], 1.0f));
    }
    aT[m * 256 + k] = (_Float16)v;
  }

  v8f c = {};
  for (int c2 = 0; c2 < 2; ++c2) {
    __syncthreads();                       // prior reads of bT done (and aT staged)
    if (w == 0) {                          // one wave issues the DMA
      tdm_load_chunk(bT, wgp + (size_t)c2 * cCHUNK);
      __builtin_amdgcn_s_wait_tensorcnt(0);
    }
    __syncthreads();                       // bT visible to all waves
#pragma unroll
    for (int kt = 0; kt < 4; ++kt) {
      v16h a = frag_a(aT, 256, c2 * 128 + kt * 32, lane);
      v16h b = frag_bs(bT, kt, w, lane);
      WMMA_F16(c, a, b);
    }
  }
  {
    int n = w * 16 + (lane & 15), mb = (lane & 16) ? 8 : 0;
#pragma unroll
    for (int r = 0; r < 8; ++r) {
      int m = mb + r, gr = row0 + m;
      float upd = silu_f(c[r] + bg[n]);
      yT[m * cD + n] = h[(size_t)gr * cD + n] + upd;
    }
  }
  __syncthreads();
  if (tid < 16) {
    float s = 0.f;
    for (int n = 0; n < cD; ++n) s += yT[tid * cD + n];
    float mean = s / cD, var = 0.f;
    for (int n = 0; n < cD; ++n) { float d = yT[tid * cD + n] - mean; var += d * d; }
    mu[tid] = mean; rs[tid] = rsqrtf(var / cD + 1e-5f);
  }
  __syncthreads();
  for (int i = tid; i < 16 * cD; i += 256) {
    int m = i >> 7, n = i & 127;
    hout[(size_t)(row0 + m) * cD + n] = (yT[i] - mu[m]) * rs[m] * lg[n] + lb[n];
  }
}

// -------- mean pool over nodes --------
__global__ void k_pmean(const float* __restrict__ h, float* __restrict__ hmean) {
  int gid = blockIdx.x * blockDim.x + threadIdx.x;
  if (gid >= cB * cD) return;
  int b = gid / cD, d = gid % cD;
  float s = 0.f;
  const float* p = h + (size_t)b * cN * cD + d;
  for (int n = 0; n < cN; ++n) {
    __builtin_prefetch(p + (size_t)(n + 8) * cD, 0, 0);
    s += p[(size_t)n * cD];
  }
  hmean[gid] = s * (1.0f / cN);
}

// -------- pooled MLP head: f --------
__global__ void k_pmlp(const float* __restrict__ hmean, const float* __restrict__ Wp1,
                       const float* __restrict__ bp1, const float* __restrict__ Wp2,
                       const float* __restrict__ bp2, float* __restrict__ out) {
  __shared__ float u[cB * cD];
  int tid = threadIdx.x;
  if (tid < cB * cD) {
    int b = tid / cD, d = tid % cD;
    float acc = bp1[d];
    for (int k = 0; k < cD; ++k) acc += hmean[b * cD + k] * Wp1[k * cD + d];
    u[tid] = silu_f(acc);
  }
  __syncthreads();
  if (tid < cB) {
    float acc = bp2[0];
    for (int d = 0; d < cD; ++d) acc += u[tid * cD + d] * Wp2[d];
    out[tid] = acc;
  }
}

// -------- move decoder: diffs --------
__global__ void k_dec(const float* __restrict__ h, const int* __restrict__ mn,
                      const unsigned char* __restrict__ mask,  // jax bool -> 1 byte
                      const _Float16* __restrict__ wd1p,       // 4 swizzled chunks (TDM src)
                      const float* __restrict__ bd1,
                      const _Float16* __restrict__ wd2p,       // 1 swizzled chunk (TDM src)
                      const float* __restrict__ bd2,
                      const float* __restrict__ Wd3, const float* __restrict__ bd3,
                      float* __restrict__ out) {
  __shared__ _Float16 aT[16 * cD];   // 4 KB (one 128-wide K chunk of gathered A)
  __shared__ _Float16 bT[cCHUNK];    // 32 KB (swizzled, TDM-filled)
  __shared__ _Float16 zT[16 * cD];   // 4 KB
  int tid = threadIdx.x, lane = tid & 31, w = tid >> 5;
  int row0 = blockIdx.x * 16;        // row in [0, B*M)

  v8f c = {};
  for (int cc = 0; cc < 4; ++cc) {   // 4 gathered nodes -> K = 4*128
    __syncthreads();
    for (int i = tid; i < 16 * cD; i += 256) {
      int m = i >> 7, k = i & 127;
      int gr = row0 + m, b = gr / cM, mm = gr % cM;
      int node = mn[(b * cM + mm) * 4 + cc];
      node = node < 0 ? 0 : (node > cN - 1 ? cN - 1 : node);
      aT[i] = (_Float16)h[((size_t)b * cN + node) * cD + k];
    }
    if (w == 0) {
      tdm_load_chunk(bT, wd1p + (size_t)cc * cCHUNK);
      __builtin_amdgcn_s_wait_tensorcnt(0);
    }
    __syncthreads();
#pragma unroll
    for (int kt = 0; kt < 4; ++kt) {
      v16h a = frag_a(aT, cD, kt * 32, lane);
      v16h b = frag_bs(bT, kt, w, lane);
      WMMA_F16(c, a, b);
    }
  }
  {
    int n = w * 16 + (lane & 15), mb = (lane & 16) ? 8 : 0;
#pragma unroll
    for (int r = 0; r < 8; ++r) zT[(mb + r) * cD + n] = (_Float16)silu_f(c[r] + bd1[n]);
  }
  __syncthreads();
  if (w == 0) {
    tdm_load_chunk(bT, wd2p);
    __builtin_amdgcn_s_wait_tensorcnt(0);
  }
  __syncthreads();
  v8f c2 = {};
#pragma unroll
  for (int kt = 0; kt < 4; ++kt) {
    v16h a = frag_a(zT, cD, kt * 32, lane);
    v16h b = frag_bs(bT, kt, w, lane);
    WMMA_F16(c2, a, b);
  }
  __syncthreads();
  {
    int n = w * 16 + (lane & 15), mb = (lane & 16) ? 8 : 0;
#pragma unroll
    for (int r = 0; r < 8; ++r) zT[(mb + r) * cD + n] = (_Float16)silu_f(c2[r] + bd2[n]);
  }
  __syncthreads();
  if (tid < 16) {
    float acc = bd3[0];
    for (int n = 0; n < cD; ++n) acc += (float)zT[tid * cD + n] * Wd3[n];
    int gr = row0 + tid, b = gr / cM, mm = gr % cM;
    out[cB + gr] = mask[b * cM + mm] ? acc : -__builtin_inff();
  }
}

extern "C" void kernel_launch(void* const* d_in, const int* in_sizes, int n_in,
                              void* d_out, int out_size, void* d_ws, size_t ws_size,
                              hipStream_t stream) {
  const float* x    = (const float*)d_in[0];
  const int*   ei   = (const int*)d_in[1];
  const int*   mn   = (const int*)d_in[2];
  const unsigned char* mmask = (const unsigned char*)d_in[3];  // bool dtype
  const float* t    = (const float*)d_in[4];
  const float* Wt1  = (const float*)d_in[5];
  const float* bt1  = (const float*)d_in[6];
  const float* Wt2  = (const float*)d_in[7];
  const float* bt2  = (const float*)d_in[8];
  const float* We1  = (const float*)d_in[9];
  const float* be1  = (const float*)d_in[10];
  const float* We2  = (const float*)d_in[11];
  const float* be2  = (const float*)d_in[12];
  const float* Wg   = (const float*)d_in[13];
  const float* bg   = (const float*)d_in[14];
  const float* lng  = (const float*)d_in[15];
  const float* lnb  = (const float*)d_in[16];
  const float* Wp1  = (const float*)d_in[17];
  const float* bp1  = (const float*)d_in[18];
  const float* Wp2  = (const float*)d_in[19];
  const float* bp2  = (const float*)d_in[20];
  const float* Wd1  = (const float*)d_in[21];
  const float* bd1  = (const float*)d_in[22];
  const float* Wd2  = (const float*)d_in[23];
  const float* bd2  = (const float*)d_in[24];
  const float* Wd3  = (const float*)d_in[25];
  const float* bd3  = (const float*)d_in[26];

  // workspace layout: ~17.4 MB total
  float* temb  = (float*)d_ws;                           // 512
  float* hbuf  = temb + 512;                             // cROWS*cD
  float* agg   = hbuf + (size_t)cROWS * cD;              // cROWS*cD
  float* deg   = agg  + (size_t)cROWS * cD;              // cN
  float* hmean = deg + cN;                               // 512
  _Float16* wpack = (_Float16*)(hmean + 512);            // 17 chunks * 16384 f16
  _Float16* wgp   = wpack;                               // 12 chunks (6 layers x 2)
  _Float16* wd1p  = wpack + (size_t)12 * cCHUNK;         // 4 chunks
  _Float16* wd2p  = wpack + (size_t)16 * cCHUNK;         // 1 chunk
  float* out   = (float*)d_out;

  k_temb<<<1, 512, 0, stream>>>(t, Wt1, bt1, Wt2, bt2, temb);
  k_prep<<<(17 * cCHUNK) / 256, 256, 0, stream>>>(Wg, Wd1, Wd2, wpack);
  k_encoder<<<cROWS / 16, 256, 0, stream>>>(x, We1, be1, We2, be2, temb, hbuf);

  hipMemsetAsync(deg, 0, cN * sizeof(float), stream);
  k_deg<<<cE / 256, 256, 0, stream>>>(ei, deg);

  for (int i = 0; i < cL; ++i) {
    hipMemsetAsync(agg, 0, (size_t)cROWS * cD * sizeof(float), stream);
    k_scatter<<<(cE * 32) / 256, 256, 0, stream>>>(ei, hbuf, agg);
    k_layer<<<cROWS / 16, 256, 0, stream>>>(hbuf, agg, deg,
                                            wgp + (size_t)i * 2 * cCHUNK,
                                            bg + i * cD, lng + i * cD, lnb + i * cD, hbuf);
  }

  k_pmean<<<(cB * cD + 255) / 256, 256, 0, stream>>>(hbuf, hmean);
  k_pmlp<<<1, 512, 0, stream>>>(hmean, Wp1, bp1, Wp2, bp2, out);
  k_dec<<<(cB * cM) / 16, 256, 0, stream>>>(hbuf, mn, mmask, wd1p, bd1, wd2p, bd2, Wd3, bd3, out);
}
